// UnidirectionalAttn_7301444403621
// MI455X (gfx1250) — compile-verified, matrix operand order checked by reading, not run on hardware
//
#include <hip/hip_runtime.h>

typedef __attribute__((ext_vector_type(16))) _Float16 v16h;
typedef __attribute__((ext_vector_type(8)))  float    v8f;
typedef unsigned int v4u __attribute__((ext_vector_type(4)));
typedef unsigned int v8u __attribute__((ext_vector_type(8)));

#define BATCH 4
#define SEQ   2048
#define HID   1024
#define NHEAD 16
#define HD    64

#define BM 128
#define BN 128
#define BK 64
#define LDT 72   // LDS row stride in halves = 64 data + 4-dword TDM pad

// ---------------------------------------------------------------------------
// Tensor Data Mover: load a 2D f16 tile (tile_x elems per row, tile_y rows,
// row stride stride_elems) from global into LDS with +4-dword padding every
// 32 dwords (=> LDS row stride 72 halves). One scalar instruction per tile.
// D# layout per CDNA5 ISA ch.8: group0 = {count, lds_addr, global_addr, type},
// group1 = {data_size/pad, tensor dims, tile dims, dim0 stride}.
// ---------------------------------------------------------------------------
__device__ __forceinline__ void tdm_load_tile_f16(const void* gsrc, const void* lds_dst,
                                                  unsigned tile_x, unsigned tile_y,
                                                  unsigned stride_elems) {
  union { v4u v; unsigned u[4]; } g0;
  union { v8u v; unsigned u[8]; } g1;
  unsigned long long ga = (unsigned long long)gsrc;
  unsigned lo = (unsigned)(unsigned long long)lds_dst;  // low 32 bits of generic ptr = LDS byte offset
  g0.u[0] = 1u;                                   // count=1, user mode, no gather
  g0.u[1] = lo;                                   // lds_addr
  g0.u[2] = (unsigned)ga;                         // global_addr[31:0]
  g0.u[3] = (unsigned)(ga >> 32) | (2u << 30);    // global_addr[56:32] | type=2
  // data_size=1 (2B), pad_enable=1, pad_interval=4 (32 dwords), pad_amount=3 (4 dwords)
  g1.u[0] = (1u << 16) | (1u << 20) | (4u << 22) | (3u << 25);
  g1.u[1] = (tile_x & 0xFFFFu) << 16;                       // tensor_dim0 = tile_x
  g1.u[2] = (tile_x >> 16) | ((tile_y & 0xFFFFu) << 16);    // tensor_dim0 hi | tensor_dim1 = tile_y
  g1.u[3] = (tile_y >> 16) | ((tile_x & 0xFFFFu) << 16);    // tensor_dim1 hi | tile_dim0
  g1.u[4] = tile_y & 0xFFFFu;                               // tile_dim1 (tile_dim2 = 0)
  g1.u[5] = stride_elems;                                   // tensor_dim0_stride[31:0]
  g1.u[6] = 0;
  g1.u[7] = 0;
  asm volatile("tensor_load_to_lds %0, %1" :: "s"(g0.v), "s"(g1.v) : "memory");
}

// ---------------------------------------------------------------------------
// Fragment loader for v_wmma_f32_16x16x32_f16 operands (A: lane holds row
// M=lane%16; half h -> K = h%8 + 16*(h/8) + 8*(lane/16); B staged transposed
// in LDS so the same loader serves both).
// ---------------------------------------------------------------------------
__device__ __forceinline__ v16h lds_frag(const _Float16* __restrict__ tile,
                                         int row0, int col0, int stride, int lane) {
  const int g = lane >> 4;
  const _Float16* p = tile + (size_t)(row0 + (lane & 15)) * stride + col0 + g * 8;
  union { v16h v; uint2 q[4]; } u;
  u.q[0] = *(const uint2*)(p + 0);
  u.q[1] = *(const uint2*)(p + 4);
  u.q[2] = *(const uint2*)(p + 16);
  u.q[3] = *(const uint2*)(p + 20);
  return u.v;
}

__device__ __forceinline__ v8f vzero8() {
  v8f z;
#pragma unroll
  for (int i = 0; i < 8; ++i) z[i] = 0.0f;
  return z;
}

__device__ __forceinline__ v8f wmma16(v16h a, v16h b, v8f c) {
  return __builtin_amdgcn_wmma_f32_16x16x32_f16(false, a, false, b, (short)0, c, false, false);
}

// ---------------------------------------------------------------------------
// Kernel 0: X = f16(x), Xp = f16(x + pos)    (q+pos_q == (x+pos)@Wq etc.)
// ---------------------------------------------------------------------------
__global__ __launch_bounds__(256)
void prep_kernel(const float* __restrict__ x, const float* __restrict__ pos,
                 _Float16* __restrict__ Xf, _Float16* __restrict__ Xp) {
  int i = blockIdx.x * 256 + threadIdx.x;
  float xv = x[i];
  Xf[i] = (_Float16)xv;
  Xp[i] = (_Float16)(xv + pos[i & (SEQ * HID - 1)]);
}

// Kernel 0b: WT[n][k] = f16(W[k][n])  (pre-transpose weights for TDM B tiles)
__global__ __launch_bounds__(256)
void wtrans_kernel(const float* __restrict__ W, _Float16* __restrict__ WT, int K, int N) {
  int n = blockIdx.x * 256 + threadIdx.x;   // coalesced read along N
  int k = blockIdx.y;
  WT[(size_t)n * K + k] = (_Float16)W[(size_t)k * N + n];
}

// ---------------------------------------------------------------------------
// Tiled WMMA GEMM: C[M,N] = A[M,K](f16) * WT[N,K](f16)^T
// 128x128 block tile, K-tile 64, double-buffered TDM tile loads.
// mode 0: QKV epilogue (A = Xp for q/k thirds, X for v third; V written
//         transposed per head for TDM-friendly attention loads)
// mode 1: f32 store to Co.
// ---------------------------------------------------------------------------
__global__ __launch_bounds__(256)
void gemm_kernel(const _Float16* __restrict__ A0, const _Float16* __restrict__ A1,
                 const _Float16* __restrict__ WT, int M, int N, int K,
                 _Float16* __restrict__ Qo, _Float16* __restrict__ Ko,
                 _Float16* __restrict__ Vt, float* __restrict__ Co, int mode) {
  __shared__ _Float16 sA[2][BM * LDT];
  __shared__ _Float16 sB[2][BN * LDT];
  (void)M;

  const int tid  = threadIdx.x;
  const int lane = tid & 31;
  const int wave = tid >> 5;
  const int n0 = blockIdx.x * BN;
  const int m0 = blockIdx.y * BM;
  const _Float16* __restrict__ A = (mode == 0 && n0 >= 2 * HID) ? A1 : A0;
  const _Float16* __restrict__ Arow = A + (size_t)m0 * K;
  const _Float16* __restrict__ Brow = WT + (size_t)n0 * K;

  const int wm = (wave >> 2) * 64;
  const int wn = (wave & 3) * 32;

  v8f acc[4][2];
#pragma unroll
  for (int i = 0; i < 4; ++i)
#pragma unroll
    for (int j = 0; j < 2; ++j) acc[i][j] = vzero8();

  if (wave == 0) {
    tdm_load_tile_f16(Arow, sA[0], BK, BM, K);
    tdm_load_tile_f16(Brow, sB[0], BK, BN, K);
  }

  const int nIter = K / BK;
  for (int it = 0; it < nIter; ++it) {
    const int buf = it & 1;
    if (wave == 0) {
      if (it + 1 < nIter) {
        tdm_load_tile_f16(Arow + (it + 1) * BK, sA[buf ^ 1], BK, BM, K);
        tdm_load_tile_f16(Brow + (it + 1) * BK, sB[buf ^ 1], BK, BN, K);
        __builtin_amdgcn_s_wait_tensorcnt(2);   // current buf's 2 loads retired
      } else {
        __builtin_amdgcn_s_wait_tensorcnt(0);
      }
    }
    __syncthreads();

#pragma unroll
    for (int ks = 0; ks < BK; ks += 32) {
      v16h af[4], bf[2];
#pragma unroll
      for (int i = 0; i < 4; ++i) af[i] = lds_frag(sA[buf], wm + i * 16, ks, LDT, lane);
#pragma unroll
      for (int j = 0; j < 2; ++j) bf[j] = lds_frag(sB[buf], wn + j * 16, ks, LDT, lane);
#pragma unroll
      for (int i = 0; i < 4; ++i)
#pragma unroll
        for (int j = 0; j < 2; ++j) acc[i][j] = wmma16(af[i], bf[j], acc[i][j]);
    }
    __syncthreads();   // all waves done reading buf before it is refilled
  }

  const int lrow = (lane >> 4) * 8;
  const int lcol = lane & 15;
#pragma unroll
  for (int i = 0; i < 4; ++i)
#pragma unroll
    for (int j = 0; j < 2; ++j)
#pragma unroll
      for (int r = 0; r < 8; ++r) {
        int mg = m0 + wm + i * 16 + lrow + r;
        int ng = n0 + wn + j * 16 + lcol;
        float v = acc[i][j][r];
        if (mode == 1) {
          Co[(size_t)mg * N + ng] = v;
        } else {
          int b = mg >> 11, s = mg & 2047;
          int kind = ng >> 10, c3 = ng & 1023;
          int head = c3 >> 6, hh = c3 & 63;
          if (kind == 0)
            Qo[(((size_t)(b * NHEAD + head)) * SEQ + s) * HD + hh] = (_Float16)v;
          else if (kind == 1)
            Ko[(((size_t)(b * NHEAD + head)) * SEQ + s) * HD + hh] = (_Float16)v;
          else  // V stored transposed: [b, head, h, s]
            Vt[(((size_t)(b * NHEAD + head)) * HD + hh) * SEQ + s] = (_Float16)v;
        }
      }
}

// ---------------------------------------------------------------------------
// Kernel 2: flash attention; Q / K / V^T tiles all via TDM, K+V^T
// double-buffered over the key loop. 8 waves; wave w owns 16 query rows.
// ---------------------------------------------------------------------------
#define QT 128
#define KT 64

__global__ __launch_bounds__(256)
void attn_kernel(const _Float16* __restrict__ Q, const _Float16* __restrict__ K,
                 const _Float16* __restrict__ Vt, _Float16* __restrict__ Cmb) {
  __shared__ _Float16 sQ[QT * LDT];            // query tile
  __shared__ _Float16 sKV[2][2 * KT * LDT];    // per buffer: K tile | V^T tile
  __shared__ _Float16 sP[8 * 16 * LDT];        // per-wave probability staging

  const int tid  = threadIdx.x;
  const int lane = tid & 31;
  const int wave = tid >> 5;
  const int lrow = (lane >> 4) * 8;
  const int lcol = lane & 15;

  const int qt = blockIdx.x;
  const int bh = blockIdx.y;
  const int q0 = qt * QT;
  const _Float16* __restrict__ Qh  = Q  + (size_t)bh * SEQ * HD;
  const _Float16* __restrict__ Kh  = K  + (size_t)bh * SEQ * HD;
  const _Float16* __restrict__ Vth = Vt + (size_t)bh * HD * SEQ;   // [h][s]

  const int nkt = (q0 + QT) / KT;   // causal: key tiles covering keys < q0+128

  if (wave == 0) {
    tdm_load_tile_f16(Qh + (size_t)q0 * HD, sQ, 64, QT, HD);       // Q tile
    tdm_load_tile_f16(Kh, sKV[0], 64, KT, HD);                     // K tile 0
    tdm_load_tile_f16(Vth, sKV[0] + KT * LDT, 64, HD, SEQ);        // V^T tile 0
    __builtin_amdgcn_s_wait_tensorcnt(2);      // in-order: Q retired
  }
  __syncthreads();
  v16h aQ0 = lds_frag(sQ, wave * 16, 0, LDT, lane);
  v16h aQ1 = lds_frag(sQ, wave * 16, 32, LDT, lane);

  float m_r[8], l_r[8];
  v8f accO[4];
#pragma unroll
  for (int r = 0; r < 8; ++r) { m_r[r] = -1e30f; l_r[r] = 0.0f; }
#pragma unroll
  for (int t = 0; t < 4; ++t) accO[t] = vzero8();

  _Float16* sPw = sP + wave * 16 * LDT;

  for (int kt = 0; kt < nkt; ++kt) {
    const int buf = kt & 1;
    const int kb  = kt * KT;
    if (wave == 0) {
      if (kt + 1 < nkt) {
        const int kb1 = (kt + 1) * KT;
        tdm_load_tile_f16(Kh + (size_t)kb1 * HD, sKV[buf ^ 1], 64, KT, HD);
        tdm_load_tile_f16(Vth + kb1, sKV[buf ^ 1] + KT * LDT, 64, HD, SEQ);
        __builtin_amdgcn_s_wait_tensorcnt(2);  // current buf's K+V retired
      } else {
        __builtin_amdgcn_s_wait_tensorcnt(0);
      }
    }
    __syncthreads();
    const _Float16* sK  = sKV[buf];
    const _Float16* sVT = sKV[buf] + KT * LDT;

    // S = Q * K^T  (16 queries x 64 keys per wave)
    v8f sAcc[4];
#pragma unroll
    for (int nt = 0; nt < 4; ++nt) {
      sAcc[nt] = vzero8();
      v16h b0 = lds_frag(sK, nt * 16, 0, LDT, lane);
      v16h b1 = lds_frag(sK, nt * 16, 32, LDT, lane);
      sAcc[nt] = wmma16(aQ0, b0, sAcc[nt]);
      sAcc[nt] = wmma16(aQ1, b1, sAcc[nt]);
    }

    // scale + causal mask + online softmax (row lives across 16 lanes)
#pragma unroll
    for (int r = 0; r < 8; ++r) {
      int qg = q0 + wave * 16 + lrow + r;
      float mt = -1e30f;
#pragma unroll
      for (int nt = 0; nt < 4; ++nt) {
        int kg = kb + nt * 16 + lcol;
        float v = sAcc[nt][r] * 0.125f;        // 1/sqrt(64)
        if (kg > qg) v = -10000.0f;            // reference NEG_INF replace
        sAcc[nt][r] = v;
        mt = fmaxf(mt, v);
      }
#pragma unroll
      for (int off = 1; off < 16; off <<= 1) mt = fmaxf(mt, __shfl_xor(mt, off, 32));
      float mnew  = fmaxf(m_r[r], mt);
      float alpha = __expf(m_r[r] - mnew);
      float rs = 0.0f;
#pragma unroll
      for (int nt = 0; nt < 4; ++nt) {
        float p = __expf(sAcc[nt][r] - mnew);
        sAcc[nt][r] = p;
        rs += p;
      }
#pragma unroll
      for (int off = 1; off < 16; off <<= 1) rs += __shfl_xor(rs, off, 32);
      l_r[r] = l_r[r] * alpha + rs;
      m_r[r] = mnew;
#pragma unroll
      for (int ht = 0; ht < 4; ++ht) accO[ht][r] *= alpha;
    }

    // stage P (16x64 f16) through LDS: C-layout -> A-fragment layout
#pragma unroll
    for (int nt = 0; nt < 4; ++nt)
#pragma unroll
      for (int r = 0; r < 8; ++r)
        sPw[(lrow + r) * LDT + nt * 16 + lcol] = (_Float16)sAcc[nt][r];
    __syncthreads();

    v16h aP0 = lds_frag(sPw, 0, 0, LDT, lane);
    v16h aP1 = lds_frag(sPw, 0, 32, LDT, lane);
#pragma unroll
    for (int ht = 0; ht < 4; ++ht) {
      v16h b0 = lds_frag(sVT, ht * 16, 0, LDT, lane);
      v16h b1 = lds_frag(sVT, ht * 16, 32, LDT, lane);
      accO[ht] = wmma16(aP0, b0, accO[ht]);
      accO[ht] = wmma16(aP1, b1, accO[ht]);
    }
    __syncthreads();   // done reading buf before wave0 refills it
  }

  // combined[b, s, head*64 + h] = O / l
  const int b = bh >> 4, head = bh & 15;
#pragma unroll
  for (int r = 0; r < 8; ++r) {
    float inv = 1.0f / l_r[r];
    int qg = q0 + wave * 16 + lrow + r;
    size_t rowoff = ((size_t)b * SEQ + qg) * HID + head * HD;
#pragma unroll
    for (int ht = 0; ht < 4; ++ht)
      Cmb[rowoff + ht * 16 + lcol] = (_Float16)(accO[ht][r] * inv);
  }
}

// ---------------------------------------------------------------------------
extern "C" void kernel_launch(void* const* d_in, const int* in_sizes, int n_in,
                              void* d_out, int out_size, void* d_ws, size_t ws_size,
                              hipStream_t stream) {
  (void)in_sizes; (void)n_in; (void)out_size; (void)ws_size;
  const float* x    = (const float*)d_in[0];
  const float* pos  = (const float*)d_in[1];
  const float* Wqkv = (const float*)d_in[2];
  const float* Wo   = (const float*)d_in[3];
  float* out = (float*)d_out;

  const size_t TOK = (size_t)BATCH * SEQ;   // 8192
  const size_t NEL = TOK * HID;             // 8,388,608

  char* w = (char*)d_ws;
  _Float16* Xp    = (_Float16*)w; w += NEL * 2;            // f16(x + pos)
  _Float16* Xf    = (_Float16*)w; w += NEL * 2;            // f16(x)
  _Float16* Qb    = (_Float16*)w; w += NEL * 2;            // [b, head, s, 64]
  _Float16* Kb    = (_Float16*)w; w += NEL * 2;
  _Float16* Vtb   = (_Float16*)w; w += NEL * 2;            // [b, head, 64, s]
  _Float16* Cmb   = (_Float16*)w; w += NEL * 2;            // [b, s, hid]
  _Float16* WqkvT = (_Float16*)w; w += (size_t)3 * HID * HID * 2;  // [3072][1024]
  _Float16* WoT   = (_Float16*)w; w += (size_t)HID * HID * 2;      // [1024][1024]

  prep_kernel<<<dim3((unsigned)(NEL / 256)), 256, 0, stream>>>(x, pos, Xf, Xp);
  wtrans_kernel<<<dim3(3 * HID / 256, HID), 256, 0, stream>>>(Wqkv, WqkvT, HID, 3 * HID);
  wtrans_kernel<<<dim3(HID / 256, HID), 256, 0, stream>>>(Wo, WoT, HID, HID);

  gemm_kernel<<<dim3(3 * HID / BN, (unsigned)(TOK / BM)), 256, 0, stream>>>(
      Xp, Xf, WqkvT, (int)TOK, 3 * HID, HID, Qb, Kb, Vtb, nullptr, 0);

  attn_kernel<<<dim3(SEQ / QT, BATCH * NHEAD), 256, 0, stream>>>(Qb, Kb, Vtb, Cmb);

  gemm_kernel<<<dim3(HID / BN, (unsigned)(TOK / BM)), 256, 0, stream>>>(
      Cmb, Cmb, WoT, (int)TOK, HID, HID, nullptr, nullptr, nullptr, out, 1);
}
